// RNNRegressor_68169720922492
// MI455X (gfx1250) — matrix operand order repaired and checked by reading, not compiled, not run on hardware
//
#include <hip/hip_runtime.h>
#include <hip/hip_bf16.h>

typedef _Float16 f16_t;
typedef __attribute__((ext_vector_type(16))) _Float16 v16h;
typedef __attribute__((ext_vector_type(8)))  _Float16 v8h;
typedef __attribute__((ext_vector_type(8)))  float    v8f;

#define B_SZ 1024
#define T_SZ 512
#define F_SZ 32
#define H_SZ 128
#define E_SZ 16
#define G3   384   /* 3*H */

// Branch-free activations on the TRANS32 pipe (no IEEE div sequence, no EXEC
// divergence inside the scan loop -> WMMAs schedule with EXEC all-ones).
__device__ __forceinline__ float fast_rcp(float x) {
    return __builtin_amdgcn_rcpf(x);          // v_rcp_f32
}
__device__ __forceinline__ float sigmoidf_(float x) {
    return fast_rcp(1.0f + __expf(-x));       // v_exp_f32 + v_rcp_f32
}
__device__ __forceinline__ float tanhf_(float x) {
#if __has_builtin(__builtin_amdgcn_tanhf)
    return __builtin_amdgcn_tanhf(x);         // hardware v_tanh_f32
#else
    // 1 - 2/(1+exp(2x)); saturates correctly at +/-1 via exp->inf/0.
    return 1.0f - 2.0f * fast_rcp(1.0f + __expf(2.0f * x));
#endif
}

__device__ __forceinline__ v8f wmma_f16(v16h a, v16h b, v8f c) {
    // D = A(16x32 f16) x B(32x16 f16) + C(16x16 f32)
    return __builtin_amdgcn_wmma_f32_16x16x32_f16(
        /*neg_a=*/false, a, /*neg_b=*/false, b,
        /*c_mod=*/(short)0, c, /*reuse_a=*/false, /*reuse_b=*/false);
}

// B fragment of W^T for v_wmma_f32_16x16x32_f16.
// W row-major [G][ldk] f16; B[k][n] = W[nbase+n][kc*32+k].
// Lane l: n = l%16 ; k = kc*32 + (l/16)*16 + h  (h = half index 0..15)
// -> 32 contiguous bytes of one weight row. No transpose needed.
__device__ __forceinline__ v16h load_bfrag(const f16_t* __restrict__ W, int ldk,
                                           int nbase, int kc, int lane) {
    const f16_t* p = W + (size_t)(nbase + (lane & 15)) * ldk + kc * 32 + ((lane >> 4) << 4);
    return *(const v16h*)p;
}

// A fragment (16x32 f16). row_k0 = &act[row m = lane%16][k base of this 32-chunk].
// halves 0..15 -> k = (l/16)*16 + 0..15
__device__ __forceinline__ v16h make_afrag(const f16_t* __restrict__ row_k0, int lane) {
    int k0 = (lane >> 4) << 4;
    v8h lo = *(const v8h*)(row_k0 + k0);
    v8h hi = *(const v8h*)(row_k0 + k0 + 8);
    return __builtin_shufflevector(lo, hi, 0,1,2,3,4,5,6,7,8,9,10,11,12,13,14,15);
}

// ---------------- prep kernels ----------------

__global__ void k_cvt_f16(const float* __restrict__ src, f16_t* __restrict__ dst, int n) {
    int i = blockIdx.x * blockDim.x + threadIdx.x;
    int stride = gridDim.x * blockDim.x;
    for (; i < n; i += stride) dst[i] = (f16_t)src[i];
}

__global__ void k_cvt_wih0x(const float* __restrict__ W_ih0, f16_t* __restrict__ dst) {
    // W_ih0: [384][48]; keep x columns 0..31 -> dst [384][32]
    int i = blockIdx.x * blockDim.x + threadIdx.x;
    if (i < G3 * F_SZ) {
        int g = i >> 5, k = i & 31;
        dst[i] = (f16_t)W_ih0[g * (F_SZ + E_SZ) + k];
    }
}

// cA0[b][g] = b_ih0[g] + (g<2H ? b_hh0[g] : 0) + sum_e embed[ticker[b]][e]*W_ih0[g][32+e]
// (t-invariant part of the layer-0 gate pre-activations; b_hh_n kept separate
//  because the n-gate multiplies gh_n by r)
__global__ void k_cA0(const float* __restrict__ W_ih0, const float* __restrict__ b_ih0,
                      const float* __restrict__ b_hh0, const float* __restrict__ embed,
                      const int* __restrict__ ticker, float* __restrict__ cA0) {
    int i = blockIdx.x * blockDim.x + threadIdx.x;
    if (i >= B_SZ * G3) return;
    int b = i / G3, g = i % G3;
    const float* e = embed + (size_t)ticker[b] * E_SZ;
    const float* w = W_ih0 + (size_t)g * (F_SZ + E_SZ) + F_SZ;
    float s = b_ih0[g] + (g < 2 * H_SZ ? b_hh0[g] : 0.0f);
    #pragma unroll
    for (int k = 0; k < E_SZ; ++k) s += e[k] * w[k];
    cA0[i] = s;
}

// ---------------- layer 0 scan ----------------
// 64 blocks x 256 threads; block owns 16 batch rows for all T steps.
__global__ __launch_bounds__(256) void k_scan0(
    const f16_t* __restrict__ x16,      // [B][T][32]
    const f16_t* __restrict__ wih0x,    // [384][32]
    const f16_t* __restrict__ whh0,     // [384][128]
    const float* __restrict__ cA0,      // [B][384]
    const float* __restrict__ b_hh0,    // [384]
    f16_t* __restrict__ h0a)            // [B][T][128]
{
    __shared__ __align__(16) f16_t h16[16 * H_SZ];
    const int tid  = threadIdx.x;
    const int w    = tid >> 5;
    const int lane = tid & 31;
    const int b0   = blockIdx.x * 16;
    const int m    = (lane >> 4) << 3;  // C-frag row base
    const int nc   = lane & 15;         // C-frag column
    const int gr = 16 * w, gz = H_SZ + 16 * w, gn = 2 * H_SZ + 16 * w;

    // Recurrent-weight B fragments: held in VGPRs for the whole scan.
    v16h Br[4], Bz[4], Bn[4];
    #pragma unroll
    for (int kc = 0; kc < 4; ++kc) {
        Br[kc] = load_bfrag(whh0, H_SZ, gr, kc, lane);
        Bz[kc] = load_bfrag(whh0, H_SZ, gz, kc, lane);
        Bn[kc] = load_bfrag(whh0, H_SZ, gn, kc, lane);
    }
    // Input-weight fragments for the x part (K = 32 exactly: one WMMA per gate).
    v16h BxR = load_bfrag(wih0x, F_SZ, gr, 0, lane);
    v16h BxZ = load_bfrag(wih0x, F_SZ, gz, 0, lane);
    v16h BxN = load_bfrag(wih0x, F_SZ, gn, 0, lane);

    // Bias/embedding C fragments (t-invariant).
    v8f Cr, Cz, Cn, Cbn, h;
    float bnn = b_hh0[gn + nc];
    #pragma unroll
    for (int v = 0; v < 8; ++v) {
        int row = b0 + m + v;
        Cr[v]  = cA0[(size_t)row * G3 + gr + nc];
        Cz[v]  = cA0[(size_t)row * G3 + gz + nc];
        Cn[v]  = cA0[(size_t)row * G3 + gn + nc];
        Cbn[v] = bnn;
        h[v]   = 0.0f;
    }

    for (int i = tid; i < 16 * H_SZ; i += 256) h16[i] = (f16_t)0.0f;

    const f16_t* xrow = x16 + (size_t)(b0 + (lane & 15)) * T_SZ * F_SZ;
    const f16_t* hrow = h16 + (lane & 15) * H_SZ;

    for (int t = 0; t < T_SZ; ++t) {
        __syncthreads();  // h16 writes from previous step visible
        v16h Ax = make_afrag(xrow + t * F_SZ, lane);
        v16h Ah[4];
        #pragma unroll
        for (int kc = 0; kc < 4; ++kc) Ah[kc] = make_afrag(hrow + kc * 32, lane);
        if (t + 1 < T_SZ) __builtin_prefetch(xrow + (t + 1) * F_SZ, 0, 3);

        v8f ar = Cr, az = Cz, gxn = Cn, ghn = Cbn;
        ar  = wmma_f16(Ax, BxR, ar);
        az  = wmma_f16(Ax, BxZ, az);
        gxn = wmma_f16(Ax, BxN, gxn);
        #pragma unroll
        for (int kc = 0; kc < 4; ++kc) {
            ar  = wmma_f16(Ah[kc], Br[kc], ar);
            az  = wmma_f16(Ah[kc], Bz[kc], az);
            ghn = wmma_f16(Ah[kc], Bn[kc], ghn);
        }
        __syncthreads();  // every wave done reading old h16
        #pragma unroll
        for (int v = 0; v < 8; ++v) {
            float r  = sigmoidf_(ar[v]);
            float z  = sigmoidf_(az[v]);
            float n  = tanhf_(gxn[v] + r * ghn[v]);
            float hn = n + z * (h[v] - n);   // == (1-z)*n + z*h
            h[v] = hn;
            f16_t hv = (f16_t)hn;
            h16[(m + v) * H_SZ + 16 * w + nc] = hv;
            h0a[((size_t)(b0 + m + v) * T_SZ + t) * H_SZ + 16 * w + nc] = hv;
        }
    }
}

// ---------------- layer 1 scan ----------------
// Input GEMM (h0 @ W_ih1^T) fused per step -> no 768 MB gx buffer.
__global__ __launch_bounds__(256) void k_scan1(
    const f16_t* __restrict__ h0a,     // [B][T][128]
    const f16_t* __restrict__ wih1,    // [384][128]
    const f16_t* __restrict__ whh1,    // [384][128]
    const float* __restrict__ b_ih1,   // [384]
    const float* __restrict__ b_hh1,   // [384]
    float* __restrict__ hlast)         // [B][128]
{
    __shared__ __align__(16) f16_t h16[16 * H_SZ];
    const int tid  = threadIdx.x;
    const int w    = tid >> 5;
    const int lane = tid & 31;
    const int b0   = blockIdx.x * 16;
    const int m    = (lane >> 4) << 3;
    const int nc   = lane & 15;
    const int gr = 16 * w, gz = H_SZ + 16 * w, gn = 2 * H_SZ + 16 * w;

    v16h Br[4], Bz[4], Bn[4];
    #pragma unroll
    for (int kc = 0; kc < 4; ++kc) {
        Br[kc] = load_bfrag(whh1, H_SZ, gr, kc, lane);
        Bz[kc] = load_bfrag(whh1, H_SZ, gz, kc, lane);
        Bn[kc] = load_bfrag(whh1, H_SZ, gn, kc, lane);
    }

    float cr  = b_ih1[gr + nc] + b_hh1[gr + nc];
    float cz  = b_ih1[gz + nc] + b_hh1[gz + nc];
    float cn  = b_ih1[gn + nc];
    float bnn = b_hh1[gn + nc];
    v8f Cr, Cz, Cn, Cbn, h;
    #pragma unroll
    for (int v = 0; v < 8; ++v) { Cr[v] = cr; Cz[v] = cz; Cn[v] = cn; Cbn[v] = bnn; h[v] = 0.0f; }

    for (int i = tid; i < 16 * H_SZ; i += 256) h16[i] = (f16_t)0.0f;

    const f16_t* xrow = h0a + (size_t)(b0 + (lane & 15)) * T_SZ * H_SZ;
    const f16_t* hrow = h16 + (lane & 15) * H_SZ;

    for (int t = 0; t < T_SZ; ++t) {
        __syncthreads();
        v16h Ax[4], Ah[4];
        #pragma unroll
        for (int kc = 0; kc < 4; ++kc) {
            Ax[kc] = make_afrag(xrow + (size_t)t * H_SZ + kc * 32, lane);
            Ah[kc] = make_afrag(hrow + kc * 32, lane);
        }
        if (t + 1 < T_SZ) __builtin_prefetch(xrow + (size_t)(t + 1) * H_SZ, 0, 3);

        v8f ar = Cr, az = Cz, gxn = Cn, ghn = Cbn;
        #pragma unroll
        for (int kc = 0; kc < 4; ++kc) {
            // input-weight fragments reloaded per step (96 KB total; WGP$/L2 resident)
            v16h BiR = load_bfrag(wih1, H_SZ, gr, kc, lane);
            v16h BiZ = load_bfrag(wih1, H_SZ, gz, kc, lane);
            v16h BiN = load_bfrag(wih1, H_SZ, gn, kc, lane);
            ar  = wmma_f16(Ax[kc], BiR, ar);
            az  = wmma_f16(Ax[kc], BiZ, az);
            gxn = wmma_f16(Ax[kc], BiN, gxn);
            ar  = wmma_f16(Ah[kc], Br[kc], ar);
            az  = wmma_f16(Ah[kc], Bz[kc], az);
            ghn = wmma_f16(Ah[kc], Bn[kc], ghn);
        }
        __syncthreads();
        #pragma unroll
        for (int v = 0; v < 8; ++v) {
            float r  = sigmoidf_(ar[v]);
            float z  = sigmoidf_(az[v]);
            float n  = tanhf_(gxn[v] + r * ghn[v]);
            float hn = n + z * (h[v] - n);
            h[v] = hn;
            h16[(m + v) * H_SZ + 16 * w + nc] = (f16_t)hn;
        }
    }
    // final hidden state in fp32 for the head
    #pragma unroll
    for (int v = 0; v < 8; ++v)
        hlast[(size_t)(b0 + m + v) * H_SZ + 16 * w + nc] = h[v];
}

// ---------------- head ----------------
__global__ __launch_bounds__(128) void k_head(
    const float* __restrict__ hlast, const float* __restrict__ w1,
    const float* __restrict__ b1, const float* __restrict__ w2,
    const float* __restrict__ b2, float* __restrict__ out)
{
    __shared__ float ls[H_SZ];
    __shared__ float red[H_SZ];
    int b = blockIdx.x, j = threadIdx.x;
    ls[j] = hlast[(size_t)b * H_SZ + j];
    __syncthreads();
    float s = b1[j];
    #pragma unroll 4
    for (int k = 0; k < H_SZ; ++k) s += ls[k] * w1[(size_t)j * H_SZ + k];
    red[j] = fmaxf(s, 0.0f) * w2[j];
    __syncthreads();
    for (int off = 64; off > 0; off >>= 1) {
        if (j < off) red[j] += red[j + off];
        __syncthreads();
    }
    if (j == 0) out[b] = red[0] + b2[0];
}

// ---------------- launch ----------------
extern "C" void kernel_launch(void* const* d_in, const int* in_sizes, int n_in,
                              void* d_out, int out_size, void* d_ws, size_t ws_size,
                              hipStream_t stream)
{
    (void)in_sizes; (void)n_in; (void)out_size; (void)ws_size;
    const float* x     = (const float*)d_in[0];
    const int*   tick  = (const int*)  d_in[1];
    const float* embed = (const float*)d_in[2];
    const float* W_ih0 = (const float*)d_in[3];
    const float* W_hh0 = (const float*)d_in[4];
    const float* b_ih0 = (const float*)d_in[5];
    const float* b_hh0 = (const float*)d_in[6];
    const float* W_ih1 = (const float*)d_in[7];
    const float* W_hh1 = (const float*)d_in[8];
    const float* b_ih1 = (const float*)d_in[9];
    const float* b_hh1 = (const float*)d_in[10];
    const float* hw1   = (const float*)d_in[11];
    const float* hb1   = (const float*)d_in[12];
    const float* hw2   = (const float*)d_in[13];
    const float* hb2   = (const float*)d_in[14];
    float* out = (float*)d_out;

    char* ws = (char*)d_ws;
    size_t off = 0;
    auto alloc = [&](size_t bytes) -> char* {
        char* p = ws + off;
        off = (off + bytes + 255) & ~(size_t)255;
        return p;
    };
    f16_t* x16   = (f16_t*)alloc((size_t)B_SZ * T_SZ * F_SZ * sizeof(f16_t));
    f16_t* h0a   = (f16_t*)alloc((size_t)B_SZ * T_SZ * H_SZ * sizeof(f16_t));
    f16_t* wih0x = (f16_t*)alloc((size_t)G3 * F_SZ * sizeof(f16_t));
    f16_t* whh0  = (f16_t*)alloc((size_t)G3 * H_SZ * sizeof(f16_t));
    f16_t* wih1  = (f16_t*)alloc((size_t)G3 * H_SZ * sizeof(f16_t));
    f16_t* whh1  = (f16_t*)alloc((size_t)G3 * H_SZ * sizeof(f16_t));
    float* cA0   = (float*)alloc((size_t)B_SZ * G3 * sizeof(float));
    float* hlast = (float*)alloc((size_t)B_SZ * H_SZ * sizeof(float));

    int nX = B_SZ * T_SZ * F_SZ;
    k_cvt_f16<<<1024, 256, 0, stream>>>(x, x16, nX);
    k_cvt_f16<<<(G3 * H_SZ + 255) / 256, 256, 0, stream>>>(W_hh0, whh0, G3 * H_SZ);
    k_cvt_f16<<<(G3 * H_SZ + 255) / 256, 256, 0, stream>>>(W_ih1, wih1, G3 * H_SZ);
    k_cvt_f16<<<(G3 * H_SZ + 255) / 256, 256, 0, stream>>>(W_hh1, whh1, G3 * H_SZ);
    k_cvt_wih0x<<<(G3 * F_SZ + 255) / 256, 256, 0, stream>>>(W_ih0, wih0x);
    k_cA0<<<(B_SZ * G3 + 255) / 256, 256, 0, stream>>>(W_ih0, b_ih0, b_hh0, embed, tick, cA0);

    k_scan0<<<B_SZ / 16, 256, 0, stream>>>(x16, wih0x, whh0, cA0, b_hh0, h0a);
    k_scan1<<<B_SZ / 16, 256, 0, stream>>>(h0a, wih1, whh1, b_ih1, b_hh1, hlast);
    k_head<<<B_SZ, 128, 0, stream>>>(hlast, hw1, hb1, hw2, hb2, out);
}
